// DBGNN_44573170598127
// MI455X (gfx1250) — matrix operand order repaired and checked by reading, not compiled, not run on hardware
//
#include <hip/hip_runtime.h>
#include <hip/hip_bf16.h>

// ---------------------------------------------------------------------------
// DBGNN (hetero GATv2, 2 layers) for gfx1250.
// Dense GEMMs via v_wmma_f32_16x16x32_bf16 (fp32 accumulate), edge phase via
// wave-per-edge kernels with f32 global atomics.
// ---------------------------------------------------------------------------

typedef __attribute__((ext_vector_type(16))) __bf16          bf16x16;
typedef __attribute__((ext_vector_type(16))) unsigned short  ushort16;
typedef __attribute__((ext_vector_type(8)))  float           floatx8;

#define DEVINL __device__ __forceinline__

DEVINL unsigned short f32_bf16(float f) {
  unsigned u = __float_as_uint(f);
  u += 0x7FFFu + ((u >> 16) & 1u);           // round-to-nearest-even
  return (unsigned short)(u >> 16);
}

DEVINL unsigned f32_ord(float f) {           // order-preserving uint mapping
  unsigned u = __float_as_uint(f);
  return (u & 0x80000000u) ? ~u : (u | 0x80000000u);
}
DEVINL float ord_f32(unsigned o) {
  unsigned u = (o & 0x80000000u) ? (o & 0x7FFFFFFFu) : ~o;
  return __uint_as_float(u);
}

// ---------------------------------------------------------------------------
// GEMM: C[M,N] = act(A[M,K] @ B[K,N] + bias[N]); A,B f32 row-major, converted
// to bf16 in LDS. N is a template constant (64 or 128); K % 64 == 0.
// Block = 256 threads (8 waves); block tile 256 x N; wave tile 32 x N
// (two A fragments share each B fragment -> 2 WMMAs per B-frag load).
// ---------------------------------------------------------------------------
#define GEMM_BM 256
#define GEMM_BK 64

template <int N>
__global__ void __launch_bounds__(256)
dbgnn_gemm_k(const float* __restrict__ A, const float* __restrict__ B,
             const float* __restrict__ bias, float* __restrict__ C,
             int M, int K, int relu) {
  constexpr int NT = N / 16;                 // 16-wide N tiles (4 or 8)
  constexpr int NW = N / 4;                  // float4 per B row
  __shared__ unsigned short lA[GEMM_BM][GEMM_BK + 8];  // pad vs bank conflicts
  __shared__ unsigned short lB[GEMM_BK][N];

  const int tid  = threadIdx.x;
  const int wave = tid >> 5;
  const int lane = tid & 31;
  const int row0 = blockIdx.x * GEMM_BM;

  floatx8 acc0[NT], acc1[NT];
#pragma unroll
  for (int t = 0; t < NT; ++t)
#pragma unroll
    for (int r = 0; r < 8; ++r) { acc0[t][r] = 0.0f; acc1[t][r] = 0.0f; }

  for (int k0 = 0; k0 < K; k0 += GEMM_BK) {
    // ---- stage A tile (256 x 64 f32 -> bf16) ----
#pragma unroll
    for (int c = tid; c < GEMM_BM * (GEMM_BK / 4); c += 256) {
      int r  = c >> 4;
      int kk = (c & 15) << 2;
      int gr = row0 + r;
      float4 v = make_float4(0.f, 0.f, 0.f, 0.f);
      if (gr < M) {
        const float* ap = A + (size_t)gr * K + k0 + kk;
        v = *(const float4*)ap;
        if (k0 + GEMM_BK < K) __builtin_prefetch(ap + GEMM_BK, 0, 0);
      }
      lA[r][kk + 0] = f32_bf16(v.x);
      lA[r][kk + 1] = f32_bf16(v.y);
      lA[r][kk + 2] = f32_bf16(v.z);
      lA[r][kk + 3] = f32_bf16(v.w);
    }
    // ---- stage B tile (64 x N f32 -> bf16) ----
#pragma unroll
    for (int c = tid; c < GEMM_BK * NW; c += 256) {
      int r   = c / NW;
      int col = (c % NW) << 2;
      const float4 v = *(const float4*)(B + (size_t)(k0 + r) * N + col);
      lB[r][col + 0] = f32_bf16(v.x);
      lB[r][col + 1] = f32_bf16(v.y);
      lB[r][col + 2] = f32_bf16(v.z);
      lB[r][col + 3] = f32_bf16(v.w);
    }
    __syncthreads();

    const int khalf = (lane >> 4) << 3;      // 0 or 8
    const int ar    = (wave << 5) + (lane & 15);
#pragma unroll
    for (int ks = 0; ks < GEMM_BK; ks += 32) {
      // ---- two A fragments (ISA 16-bit A 16x32 layout: lane=M, split-K) ----
      ushort16 a0us, a1us;
#pragma unroll
      for (int i = 0; i < 16; ++i) {
        int k = ks + ((i < 8) ? (khalf + i) : (16 + khalf + (i - 8)));
        a0us[i] = lA[ar][k];
        a1us[i] = lA[ar + 16][k];
      }
      bf16x16 a0 = __builtin_bit_cast(bf16x16, a0us);
      bf16x16 a1 = __builtin_bit_cast(bf16x16, a1us);

      // ---- B fragments (lane=N, element index = split-K like A); each feeds 2 WMMAs ----
#pragma unroll
      for (int t = 0; t < NT; ++t) {
        ushort16 bus;
#pragma unroll
        for (int i = 0; i < 16; ++i) bus[i] = lB[ks + ((i < 8) ? (khalf + i) : (16 + khalf + (i - 8)))][(t << 4) + (lane & 15)];
        bf16x16 bf = __builtin_bit_cast(bf16x16, bus);
        acc0[t] = __builtin_amdgcn_wmma_f32_16x16x32_bf16(
            false, a0, false, bf, (short)0, acc0[t], false, false);
        acc1[t] = __builtin_amdgcn_wmma_f32_16x16x32_bf16(
            false, a1, false, bf, (short)0, acc1[t], false, false);
      }
    }
    __syncthreads();
  }

  // ---- epilogue: C/D layout M = vgpr + 8*(lane>=16), N = lane&15 ----
  const int mbase = row0 + (wave << 5) + ((lane >> 4) << 3);
  const int nlane = lane & 15;
#pragma unroll
  for (int t = 0; t < NT; ++t) {
    int coln = (t << 4) + nlane;
    float bv = bias[coln];
#pragma unroll
    for (int r = 0; r < 8; ++r) {
      int gr0 = mbase + r;
      int gr1 = gr0 + 16;
      if (gr0 < M) {
        float v = acc0[t][r] + bv;
        if (relu) v = v > 0.f ? v : 0.f;
        C[(size_t)gr0 * N + coln] = v;
      }
      if (gr1 < M) {
        float v = acc1[t][r] + bv;
        if (relu) v = v > 0.f ? v : 0.f;
        C[(size_t)gr1 * N + coln] = v;
      }
    }
  }
}

// ---------------------------------------------------------------------------
// Fill kernels
// ---------------------------------------------------------------------------
__global__ void __launch_bounds__(256)
dbgnn_fill_f32(float* p, float v, int n) {
  int i = blockIdx.x * 256 + threadIdx.x;
  if (i < n) p[i] = v;
}
__global__ void __launch_bounds__(256)
dbgnn_fill_u32(unsigned* p, unsigned v, int n) {
  int i = blockIdx.x * 256 + threadIdx.x;
  if (i < n) p[i] = v;
}

// ---------------------------------------------------------------------------
// GATv2 edge phase. D = 128 fixed (4 floats / lane, one wave / edge).
// ---------------------------------------------------------------------------
__global__ void __launch_bounds__(256)
dbgnn_edge_logits_k(const float* __restrict__ xl, const float* __restrict__ xr,
                    const int* __restrict__ src, const int* __restrict__ dst,
                    const float* __restrict__ att, float* __restrict__ logits,
                    unsigned* __restrict__ mord, int E) {
  int e = blockIdx.x * 8 + (threadIdx.x >> 5);
  if (e >= E) return;
  int lane = threadIdx.x & 31;
  int s = src[e], d = dst[e];
  float4 a = *(const float4*)(xl + (size_t)s * 128 + lane * 4);
  float4 b = *(const float4*)(xr + (size_t)d * 128 + lane * 4);
  float4 w = *(const float4*)(att + lane * 4);
  float p = 0.f, h;
  h = a.x + b.x; h = h > 0.f ? h : 0.2f * h; p += h * w.x;
  h = a.y + b.y; h = h > 0.f ? h : 0.2f * h; p += h * w.y;
  h = a.z + b.z; h = h > 0.f ? h : 0.2f * h; p += h * w.z;
  h = a.w + b.w; h = h > 0.f ? h : 0.2f * h; p += h * w.w;
#pragma unroll
  for (int off = 16; off; off >>= 1) p += __shfl_xor(p, off, 32);
  if (lane == 0) {
    logits[e] = p;
    atomicMax(mord + d, f32_ord(p));
  }
}

__global__ void __launch_bounds__(256)
dbgnn_edge_exp_k(const float* __restrict__ logits, const int* __restrict__ dst,
                 const unsigned* __restrict__ mord, float* __restrict__ evb,
                 float* __restrict__ denom, float* __restrict__ cnt, int E) {
  int e = blockIdx.x * 256 + threadIdx.x;
  if (e >= E) return;
  int d = dst[e];
  float ev = __expf(logits[e] - ord_f32(mord[d]));
  evb[e] = ev;
  atomicAdd(denom + d, ev);
  atomicAdd(cnt + d, 1.0f);
}

__global__ void __launch_bounds__(256)
dbgnn_edge_accum_k(const float* __restrict__ xl, const int* __restrict__ src,
                   const int* __restrict__ dst, const float* __restrict__ evb,
                   const float* __restrict__ denom, float* __restrict__ out, int E) {
  int e = blockIdx.x * 8 + (threadIdx.x >> 5);
  if (e >= E) return;
  int lane = threadIdx.x & 31;
  int s = src[e], d = dst[e];
  float alpha = evb[e] / denom[d];
  float4 v = *(const float4*)(xl + (size_t)s * 128 + lane * 4);
  float* o = out + (size_t)d * 128 + lane * 4;
  atomicAdd(o + 0, alpha * v.x);
  atomicAdd(o + 1, alpha * v.y);
  atomicAdd(o + 2, alpha * v.z);
  atomicAdd(o + 3, alpha * v.w);
}

__global__ void __launch_bounds__(256)
dbgnn_gat_final_k(float* __restrict__ out, const float* __restrict__ cnt,
                  const float* __restrict__ bias, int n) {
  int i = blockIdx.x * 256 + threadIdx.x;
  if (i < n * 128) {
    int node = i >> 7, d = i & 127;
    float c = cnt[node]; c = c > 1.f ? c : 1.f;
    out[i] = out[i] / c + bias[d];
  }
}

// ---------------------------------------------------------------------------
// Output head: [M,128] @ W[128,8] + b, row softmax. One wave per node:
// lane -> (o = lane&7 output, q = lane>>3 K-quarter), shuffle reductions.
// ---------------------------------------------------------------------------
__global__ void __launch_bounds__(256)
dbgnn_out_softmax_k(const float* __restrict__ x, const float* __restrict__ W,
                    const float* __restrict__ b, float* __restrict__ out, int M) {
  int n = blockIdx.x * 8 + (threadIdx.x >> 5);
  if (n >= M) return;
  int lane = threadIdx.x & 31;
  int o = lane & 7;
  int q = lane >> 3;
  float p = 0.f;
  const float* xr = x + (size_t)n * 128 + q * 32;
#pragma unroll
  for (int kk = 0; kk < 32; ++kk) p += xr[kk] * W[(q * 32 + kk) * 8 + o];
  p += __shfl_xor(p, 8, 32);
  p += __shfl_xor(p, 16, 32);          // every lane: full dot for its o
  float v = p + b[o];
  float m = v;
  m = fmaxf(m, __shfl_xor(m, 1, 32));
  m = fmaxf(m, __shfl_xor(m, 2, 32));
  m = fmaxf(m, __shfl_xor(m, 4, 32));
  float ev = __expf(v - m);
  float sm = ev;
  sm += __shfl_xor(sm, 1, 32);
  sm += __shfl_xor(sm, 2, 32);
  sm += __shfl_xor(sm, 4, 32);
  if (lane < 8) out[(size_t)n * 8 + o] = ev / sm;
}

// ---------------------------------------------------------------------------
// Orchestration
// ---------------------------------------------------------------------------
extern "C" void kernel_launch(void* const* d_in, const int* in_sizes, int n_in,
                              void* d_out, int out_size, void* d_ws, size_t ws_size,
                              hipStream_t stream) {
  (void)n_in; (void)out_size; (void)ws_size;

  const float* x_cust = (const float*)d_in[0];
  const float* x_ord  = (const float*)d_in[1];
  const int* src_co = (const int*)d_in[2];
  const int* dst_co = (const int*)d_in[3];
  const int* src_oc = (const int*)d_in[4];
  const int* dst_oc = (const int*)d_in[5];
  const int E  = in_sizes[2];
  const int NC = in_sizes[0] / 512;   // 8*64 cols
  const int NO = in_sizes[1] / 384;   // 6*64 cols

  // Param leaves in jax pytree order (dict keys sorted: layers, out, proj).
  // Per layer (20 leaves): co{Wl,Wr,att,bias,bl,br} oc{...} mlpC{W1,b1,W2,b2} mlpO{...}
  const float* const* P = (const float* const*)(d_in + 6);
  const float* outW   = P[40];
  const float* outB   = P[41];
  const float* projCW = P[42];
  const float* projCb = P[43];
  const float* projOW = P[44];
  const float* projOb = P[45];

  // Workspace carve-up
  char* w = (char*)d_ws;
  size_t off = 0;
  auto carve = [&](size_t bytes) -> char* {
    char* p = w + off;
    off += (bytes + 255) & ~(size_t)255;
    return p;
  };
  float*    xc     = (float*)carve((size_t)NC * 128 * 4);
  float*    A1     = (float*)carve((size_t)NC * 128 * 4);   // mc
  float*    A2     = (float*)carve((size_t)NC * 128 * 4);   // xl_co / xr_oc
  float*    xo     = (float*)carve((size_t)NO * 128 * 4);
  float*    B1     = (float*)carve((size_t)NO * 128 * 4);   // mo
  float*    B2     = (float*)carve((size_t)NO * 128 * 4);   // hidden / xr_co / xl_oc
  float*    logits = (float*)carve((size_t)E * 4);
  float*    evb    = (float*)carve((size_t)E * 4);
  unsigned* mord   = (unsigned*)carve((size_t)NO * 4);
  float*    denom  = (float*)carve((size_t)NO * 4);
  float*    cntb   = (float*)carve((size_t)NO * 4);

  auto gemm128 = [&](const float* A, const float* Bm, const float* bias, float* C,
                     int M, int K, int relu) {
    dbgnn_gemm_k<128><<<dim3((M + GEMM_BM - 1) / GEMM_BM), dim3(256), 0, stream>>>(
        A, Bm, bias, C, M, K, relu);
  };
  auto gemm64 = [&](const float* A, const float* Bm, const float* bias, float* C,
                    int M, int K, int relu) {
    dbgnn_gemm_k<64><<<dim3((M + GEMM_BM - 1) / GEMM_BM), dim3(256), 0, stream>>>(
        A, Bm, bias, C, M, K, relu);
  };

  auto gat = [&](const float* xl, const float* xr, const int* src, const int* dst,
                 const float* att, const float* bias, int Nd, float* outb) {
    int gn = (Nd + 255) / 256;
    dbgnn_fill_u32<<<gn, 256, 0, stream>>>(mord, 0u, Nd);
    dbgnn_fill_f32<<<gn, 256, 0, stream>>>(denom, 0.f, Nd);
    dbgnn_fill_f32<<<gn, 256, 0, stream>>>(cntb, 0.f, Nd);
    int ne = Nd * 128;
    dbgnn_fill_f32<<<(ne + 255) / 256, 256, 0, stream>>>(outb, 0.f, ne);
    int eb = (E + 7) / 8;
    dbgnn_edge_logits_k<<<eb, 256, 0, stream>>>(xl, xr, src, dst, att, logits, mord, E);
    dbgnn_edge_exp_k<<<(E + 255) / 256, 256, 0, stream>>>(logits, dst, mord, evb, denom, cntb, E);
    dbgnn_edge_accum_k<<<eb, 256, 0, stream>>>(xl, src, dst, evb, denom, outb, E);
    dbgnn_gat_final_k<<<(ne + 255) / 256, 256, 0, stream>>>(outb, cntb, bias, Nd);
  };

  // Input projections
  gemm128(x_cust, projCW, projCb, xc, NC, 512, 0);
  gemm128(x_ord,  projOW, projOb, xo, NO, 384, 0);

  for (int l = 0; l < 2; ++l) {
    const float* const* L = P + l * 20;
    float* hid = B2;                               // scratch overlay
    // node MLPs
    gemm64 (xc,  L[12], L[13], hid, NC, 128, 1);
    gemm128(hid, L[14], L[15], A1,  NC,  64, 0);   // mc
    gemm64 (xo,  L[16], L[17], hid, NO, 128, 1);
    gemm128(hid, L[18], L[19], B1,  NO,  64, 0);   // mo
    // co relation: customers -> orders (dst = orders)
    gemm128(A1, L[0], L[4], A2, NC, 128, 0);       // xl_co
    gemm128(B1, L[1], L[5], B2, NO, 128, 0);       // xr_co
    gat(A2, B2, src_co, dst_co, L[2], L[3], NO, xo);
    // oc relation: orders -> customers (dst = customers)
    gemm128(B1, L[6],  L[10], B2, NO, 128, 0);     // xl_oc
    gemm128(A1, L[7],  L[11], A2, NC, 128, 0);     // xr_oc
    gat(B2, A2, src_oc, dst_oc, L[8], L[9], NC, xc);
  }

  dbgnn_out_softmax_k<<<(NC + 7) / 8, 256, 0, stream>>>(xc, outW, outB, (float*)d_out, NC);
}
